// ClusterAwareAttention_14224931684549
// MI455X (gfx1250) — matrix operand order repaired and checked by reading, not compiled, not run on hardware
//
#include <hip/hip_runtime.h>

// ---------------- problem constants ----------------
#define BB   4
#define NN   16384
#define CC   256
#define HH   8
#define DD   32
#define KK   64
#define MTOT (BB * NN)          // 65536 rows
#define C3   (3 * CC)           // 768

typedef __attribute__((ext_vector_type(16))) __bf16 v16bf;
typedef __attribute__((ext_vector_type(8)))  __bf16 v8bf;
typedef __attribute__((ext_vector_type(8)))  float  v8f;
typedef __attribute__((ext_vector_type(4)))  int    v4i;

#define DEVINL __device__ __forceinline__

// Optional gfx1250 async global->LDS copy (guarded: absent on host pass / older clang)
#if defined(__has_builtin)
# if __has_builtin(__builtin_amdgcn_global_load_async_to_lds_b128)
#  define HAVE_ASYNC_LDS 1
# endif
#endif

DEVINL void stage16(const __bf16* g, __bf16* l) {
#ifdef HAVE_ASYNC_LDS
  typedef __attribute__((address_space(1))) v4i gv4i;
  typedef __attribute__((address_space(3))) v4i lv4i;
  __builtin_amdgcn_global_load_async_to_lds_b128((gv4i*)g, (lv4i*)l, 0, 0);
#else
  *(v8bf*)l = *(const v8bf*)g;
  *(v8bf*)(l + 8) = *(const v8bf*)(g + 8);
#endif
}
DEVINL void stage_wait() {
#ifdef HAVE_ASYNC_LDS
  asm volatile("s_wait_asynccnt 0" ::: "memory");
#endif
}

DEVINL v8f vzero() {
  v8f z;
#pragma unroll
  for (int i = 0; i < 8; ++i) z[i] = 0.0f;
  return z;
}

DEVINL v8f wmma_bf16(v16bf a, v16bf b, v8f c) {
  return __builtin_amdgcn_wmma_f32_16x16x32_bf16(false, a, false, b, (short)0, c, false, false);
}

// A fragment (16x32 bf16) from row-major bf16, vectorized: two b128 runs per lane.
// Lane L: row=row0+(L&15), K base kb=k0+16*(L>>4); frag = {K kb..kb+15}.
DEVINL v16bf load_a_bf16v(const __bf16* p, int lda, long row0, int k0, int lane) {
  const __bf16* rp = p + (row0 + (lane & 15)) * (long)lda + k0 + ((lane >> 4) << 4);
  v8bf lo = *(const v8bf*)rp;
  v8bf hi = *(const v8bf*)(rp + 8);
  v16bf a;
#pragma unroll
  for (int i = 0; i < 8; ++i) { a[i] = lo[i]; a[i + 8] = hi[i]; }
  return a;
}

// B fragment (32x16 bf16) given a pointer to this lane's 16 contiguous K values
// (column-major staging). Lane L: col=(L&15), K half = 16*(L>>4).
DEVINL v16bf load_frag16(const __bf16* p) {
  v8bf lo = *(const v8bf*)p;
  v8bf hi = *(const v8bf*)(p + 8);
  v16bf b;
#pragma unroll
  for (int i = 0; i < 8; ++i) { b[i] = lo[i]; b[i + 8] = hi[i]; }
  return b;
}

// C/D store: lane L covers col=col0+(L&15); VGPR r covers rows row0+8*(L>>4)+r.
DEVINL void store_c_bf16(__bf16* p, int ldc, long row0, int col0, int lane, v8f c) {
  int  col = col0 + (lane & 15);
  long rb  = row0 + ((lane >> 4) << 3);
#pragma unroll
  for (int r = 0; r < 8; ++r) p[(rb + r) * (long)ldc + col] = (__bf16)c[r];
}

DEVINL void store_c_f32(float* p, int ldc, long row0, int col0, int lane, v8f c, float add) {
  int  col = col0 + (lane & 15);
  long rb  = row0 + ((lane >> 4) << 3);
#pragma unroll
  for (int r = 0; r < 8; ++r) p[(rb + r) * (long)ldc + col] = c[r] + add;
}

// ---------------- kernel 1: prep (one-time bf16 conversion / transposition) ----------------
__global__ void k_prep(const float* __restrict__ A, const float* __restrict__ X,
                       const float* __restrict__ Wq, const float* __restrict__ Wp,
                       __bf16* __restrict__ Ab, __bf16* __restrict__ Xi,
                       __bf16* __restrict__ WqT, __bf16* __restrict__ WpT,
                       float* __restrict__ kcf, float* __restrict__ vcf) {
  long stride = (long)gridDim.x * blockDim.x;
  long i0 = (long)blockIdx.x * blockDim.x + threadIdx.x;
  for (long j = i0; j < (long)MTOT * CC; j += stride) Xi[j] = (__bf16)X[j];
  for (long j = i0; j < (long)MTOT * KK; j += stride) Ab[j] = (__bf16)A[j];
  for (long j = i0; j < (long)C3 * CC; j += stride) {     // WqT[col][k] = Wq[k][col]
    long col = j >> 8, k = j & 255;
    WqT[j] = (__bf16)Wq[k * C3 + col];
  }
  for (long j = i0; j < (long)CC * CC; j += stride) {     // WpT[col][k] = Wp[k][col]
    long col = j >> 8, k = j & 255;
    WpT[j] = (__bf16)Wp[k * CC + col];
  }
  for (long j = i0; j < (long)BB * KK * CC; j += stride) { kcf[j] = 0.0f; vcf[j] = 0.0f; }
}

// ---------------- kernel 2: cluster mass denom ----------------
__global__ void k_denom(const float* __restrict__ A, float* __restrict__ dinv) {
  __shared__ float red[256];
  int k = blockIdx.x, b = blockIdx.y;
  float s = 0.0f;
  for (int n = threadIdx.x; n < NN; n += 256) s += A[((long)b * NN + n) * KK + k];
  red[threadIdx.x] = s;
  __syncthreads();
  for (int o = 128; o > 0; o >>= 1) {
    if ((int)threadIdx.x < o) red[threadIdx.x] += red[threadIdx.x + o];
    __syncthreads();
  }
  if (threadIdx.x == 0) dinv[b * KK + k] = 1.0f / (red[0] + 1e-8f);
}

// ---------------- kernel 3: QKV GEMM (bf16 WMMA, 32x64 per wave) ----------------
__global__ void k_qkv(const __bf16* __restrict__ Xi, const __bf16* __restrict__ WqT,
                      __bf16* __restrict__ Qb, __bf16* __restrict__ Kb, __bf16* __restrict__ Vb) {
  __shared__ __attribute__((aligned(16))) __bf16 wt[64 * 32];  // [c][k] panel (4KB)
  int  wave = threadIdx.x >> 5, lane = threadIdx.x & 31;
  int  col0 = blockIdx.x << 6;                    // 64-col strip (12 strips)
  long row0 = (((long)blockIdx.y * 8 + wave) << 5);
  v8f acc[2][4];
#pragma unroll
  for (int i = 0; i < 2; ++i)
#pragma unroll
    for (int t = 0; t < 4; ++t) acc[i][t] = vzero();

  int sc = threadIdx.x >> 2, sp = (threadIdx.x & 3) << 3;  // staging: 16B per thread
  const __bf16* gsrc = WqT + (long)(col0 + sc) * CC + sp;

  for (int k0 = 0; k0 < CC; k0 += 32) {
    stage16(gsrc + k0, &wt[sc * 32 + sp]);        // async global->LDS (b128)
    stage_wait();
    __syncthreads();

    v16bf a0 = load_a_bf16v(Xi, CC, row0, k0, lane);
    v16bf a1 = load_a_bf16v(Xi, CC, row0 + 16, k0, lane);
#pragma unroll
    for (int t = 0; t < 4; ++t) {
      v16bf bf = load_frag16(&wt[(((t << 4) + (lane & 15)) << 5) + ((lane >> 4) << 4)]);
      acc[0][t] = wmma_bf16(a0, bf, acc[0][t]);
      acc[1][t] = wmma_bf16(a1, bf, acc[1][t]);
    }
    __syncthreads();
  }

#pragma unroll
  for (int t = 0; t < 4; ++t) {
    int gc = col0 + (t << 4);
    __bf16* dst; int c0 = gc;
    if (gc < CC)          { dst = Qb; }
    else if (gc < 2 * CC) { dst = Kb; c0 = gc - CC; }
    else                  { dst = Vb; c0 = gc - 2 * CC; }
    store_c_bf16(dst, CC, row0, c0, lane, acc[0][t]);
    store_c_bf16(dst, CC, row0 + 16, c0, lane, acc[1][t]);
  }
}

// ---------------- kernel 4: cluster pooling GEMM  Kc=A^T K, Vc=A^T V ----------------
__global__ void k_pool(const __bf16* __restrict__ Ab, const __bf16* __restrict__ Kb,
                       const __bf16* __restrict__ Vb, float* __restrict__ kcf,
                       float* __restrict__ vcf) {
  int tile = blockIdx.x;                 // 0..15: 4 cluster-row tiles x 4 col strips
  int b    = blockIdx.y;
  int wave = threadIdx.x >> 5, lane = threadIdx.x & 31;
  int row0 = (tile >> 2) << 4;           // cluster tile
  int col0 = (tile & 3) << 6;            // 64-channel strip
  const __bf16* S = wave ? Vb : Kb;
  float*        O = wave ? vcf : kcf;
  long base = (long)b * NN;
  v8f acc[4];
#pragma unroll
  for (int t = 0; t < 4; ++t) acc[t] = vzero();
  int n0 = blockIdx.z * (NN / 16);

  for (int s = 0; s < 32; ++s, n0 += 32) {
    v16bf a;   // A^T fragment: row = cluster, K-dim = n (strided gather)
    {
      int cl = row0 + (lane & 15);
      int nb = n0 + ((lane >> 4) << 4);
#pragma unroll
      for (int v = 0; v < 8; ++v) {
        int n = nb + 2 * v;
        a[2 * v]     = Ab[(base + n) * KK + cl];
        a[2 * v + 1] = Ab[(base + n + 1) * KK + cl];
      }
    }
    int nb = n0 + ((lane >> 4) << 4);
#pragma unroll
    for (int t = 0; t < 4; ++t) {
      int c = col0 + (t << 4) + (lane & 15);
      v16bf bfr;
#pragma unroll
      for (int v = 0; v < 8; ++v) {
        int n = nb + 2 * v;
        bfr[2 * v]     = S[(base + n) * (long)CC + c];
        bfr[2 * v + 1] = S[(base + n + 1) * (long)CC + c];
      }
      acc[t] = wmma_bf16(a, bfr, acc[t]);
    }
  }

  int rb = row0 + ((lane >> 4) << 3);
#pragma unroll
  for (int t = 0; t < 4; ++t) {
    int col = col0 + (t << 4) + (lane & 15);
#pragma unroll
    for (int r = 0; r < 8; ++r)
      __hip_atomic_fetch_add(&O[((long)b * KK + rb + r) * CC + col], acc[t][r],
                             __ATOMIC_RELAXED, __HIP_MEMORY_SCOPE_AGENT);
  }
}

// ---------------- kernel 5: finalize pooling ----------------
__global__ void k_finpool(const float* __restrict__ kcf, const float* __restrict__ vcf,
                          const float* __restrict__ dinv, __bf16* __restrict__ kcb,
                          __bf16* __restrict__ vcb) {
  int i  = blockIdx.x * 256 + threadIdx.x;   // 65536 = B*K*C
  int b  = i >> 14;
  int cl = (i >> 8) & 63;
  float sc = dinv[b * KK + cl];
  kcb[i] = (__bf16)(kcf[i] * sc);
  vcb[i] = (__bf16)(vcf[i] * sc);
}

// ---------------- kernel 6: fused attention ----------------
__global__ void k_attn(const __bf16* __restrict__ Qb, const __bf16* __restrict__ Ab,
                       const __bf16* __restrict__ kcb, const __bf16* __restrict__ vcb,
                       const float* __restrict__ cbias, __bf16* __restrict__ Xb) {
  __shared__ __attribute__((aligned(16))) __bf16 kc_s[KK * DD];  // [cl][d]
  __shared__ __attribute__((aligned(16))) __bf16 vc_t[DD * KK];  // [d][cl]
  __shared__ __attribute__((aligned(16))) __bf16 cb_t[KK * KK];  // [l][k]
  __shared__ __attribute__((aligned(16))) float  p_s[8][16 * KK];

  int h = blockIdx.y, b = blockIdx.z;
  int wave = threadIdx.x >> 5, lane = threadIdx.x & 31;

  for (int i = threadIdx.x; i < KK * DD; i += 256) {
    int cl = i >> 5, d = i & 31;
    kc_s[i] = kcb[((long)b * KK + cl) * CC + h * DD + d];
    int d2 = i >> 6, cl2 = i & 63;
    vc_t[i] = vcb[((long)b * KK + cl2) * CC + h * DD + d2];
  }
  for (int i = threadIdx.x; i < KK * KK; i += 256) {
    int l = i >> 6, k = i & 63;
    cb_t[i] = (__bf16)cbias[k * KK + l];
  }
  __syncthreads();

  long row0 = (long)b * NN + (((long)blockIdx.x * 8 + wave) << 4);

  v16bf qa  = load_a_bf16v(Qb, CC, row0, h * DD, lane);
  v16bf aa0 = load_a_bf16v(Ab, KK, row0, 0, lane);
  v16bf aa1 = load_a_bf16v(Ab, KK, row0, 32, lane);

  const float scale = 0.17677669529663687f;    // 1/sqrt(32)
  int cl16 = lane & 15, kh = (lane >> 4) << 4;
  v8f logit[4];
#pragma unroll
  for (int t = 0; t < 4; ++t) {
    int cl = (t << 4) + cl16;
    v16bf kb  = load_frag16(&kc_s[(cl << 5) + kh]);        // B[d][cl]=Kc[cl][d]
    v16bf cb0 = load_frag16(&cb_t[(cl << 6) + kh]);        // B[k][l], k 0..31
    v16bf cb1 = load_frag16(&cb_t[(cl << 6) + 32 + kh]);   // B[k][l], k 32..63
    v8f s    = wmma_bf16(qa, kb, vzero());
    v8f bias = wmma_bf16(aa0, cb0, vzero());
    bias     = wmma_bf16(aa1, cb1, bias);
#pragma unroll
    for (int r = 0; r < 8; ++r) logit[t][r] = s[r] * scale + bias[r];
  }

  // row softmax across 4 tiles x 16 lanes (within each half-wave)
  v8f mx;
#pragma unroll
  for (int r = 0; r < 8; ++r)
    mx[r] = fmaxf(fmaxf(logit[0][r], logit[1][r]), fmaxf(logit[2][r], logit[3][r]));
  for (int off = 1; off < 16; off <<= 1) {
#pragma unroll
    for (int r = 0; r < 8; ++r) mx[r] = fmaxf(mx[r], __shfl_xor(mx[r], off, 32));
  }
  v8f sum = vzero();
#pragma unroll
  for (int t = 0; t < 4; ++t)
#pragma unroll
    for (int r = 0; r < 8; ++r) {
      logit[t][r] = __expf(logit[t][r] - mx[r]);
      sum[r] += logit[t][r];
    }
  for (int off = 1; off < 16; off <<= 1) {
#pragma unroll
    for (int r = 0; r < 8; ++r) sum[r] += __shfl_xor(sum[r], off, 32);
  }
#pragma unroll
  for (int r = 0; r < 8; ++r) sum[r] = 1.0f / sum[r];

  // C-layout -> A-layout transpose bounce through wave-private LDS slab
  float* pw = p_s[wave];
  int half = lane >> 4;
#pragma unroll
  for (int t = 0; t < 4; ++t)
#pragma unroll
    for (int r = 0; r < 8; ++r)
      pw[(r + (half << 3)) * KK + (t << 4) + cl16] = logit[t][r] * sum[r];
  asm volatile("s_wait_dscnt 0" ::: "memory");

  v16bf pa0, pa1;
  {
    const float* rp = pw + cl16 * KK + (half << 4);
#pragma unroll
    for (int i = 0; i < 8; ++i) {
      pa0[i]     = (__bf16)rp[i];
      pa0[i + 8] = (__bf16)rp[8 + i];
      pa1[i]     = (__bf16)rp[32 + i];
      pa1[i + 8] = (__bf16)rp[40 + i];
    }
  }

#pragma unroll
  for (int ct = 0; ct < 2; ++ct) {
    int dcol = (ct << 4) + cl16;
    v16bf vb0 = load_frag16(&vc_t[(dcol << 6) + kh]);        // B[cl][d]=Vc, cl 0..31
    v16bf vb1 = load_frag16(&vc_t[(dcol << 6) + 32 + kh]);   // cl 32..63
    v8f xacc = wmma_bf16(pa0, vb0, vzero());
    xacc     = wmma_bf16(pa1, vb1, xacc);
    store_c_bf16(Xb, CC, row0, h * DD + (ct << 4), lane, xacc);
  }
}

// ---------------- kernel 7: output projection (32x64 per wave) ----------------
__global__ void k_proj(const __bf16* __restrict__ Xb, const __bf16* __restrict__ WpT,
                       const float* __restrict__ bp, float* __restrict__ out) {
  __shared__ __attribute__((aligned(16))) __bf16 wt[64 * 32];
  int  wave = threadIdx.x >> 5, lane = threadIdx.x & 31;
  int  col0 = blockIdx.x << 6;
  long row0 = (((long)blockIdx.y * 8 + wave) << 5);
  v8f acc[2][4];
#pragma unroll
  for (int i = 0; i < 2; ++i)
#pragma unroll
    for (int t = 0; t < 4; ++t) acc[i][t] = vzero();

  int sc = threadIdx.x >> 2, sp = (threadIdx.x & 3) << 3;
  const __bf16* gsrc = WpT + (long)(col0 + sc) * CC + sp;

  for (int k0 = 0; k0 < CC; k0 += 32) {
    stage16(gsrc + k0, &wt[sc * 32 + sp]);
    stage_wait();
    __syncthreads();

    v16bf a0 = load_a_bf16v(Xb, CC, row0, k0, lane);
    v16bf a1 = load_a_bf16v(Xb, CC, row0 + 16, k0, lane);
#pragma unroll
    for (int t = 0; t < 4; ++t) {
      v16bf bf = load_frag16(&wt[(((t << 4) + (lane & 15)) << 5) + ((lane >> 4) << 4)]);
      acc[0][t] = wmma_bf16(a0, bf, acc[0][t]);
      acc[1][t] = wmma_bf16(a1, bf, acc[1][t]);
    }
    __syncthreads();
  }

#pragma unroll
  for (int t = 0; t < 4; ++t) {
    float bv = bp[col0 + (t << 4) + (lane & 15)];
    store_c_f32(out, CC, row0, col0 + (t << 4), lane, acc[0][t], bv);
    store_c_f32(out, CC, row0 + 16, col0 + (t << 4), lane, acc[1][t], bv);
  }
}

// ---------------- host side ----------------
extern "C" void kernel_launch(void* const* d_in, const int* in_sizes, int n_in,
                              void* d_out, int out_size, void* d_ws, size_t ws_size,
                              hipStream_t stream) {
  const float* X     = (const float*)d_in[0];  // (B,N,C)
  const float* A     = (const float*)d_in[1];  // (B,N,K)
  const float* Wq    = (const float*)d_in[2];  // (C,3C)
  const float* Wp    = (const float*)d_in[3];  // (C,C)
  const float* bp    = (const float*)d_in[4];  // (C,)
  const float* cbias = (const float*)d_in[5];  // (K,K)
  float*       out   = (float*)d_out;          // (B,N,C)

  char* w = (char*)d_ws;
  __bf16* Xi  = (__bf16*)w; w += (size_t)MTOT * CC * 2;
  __bf16* Qb  = (__bf16*)w; w += (size_t)MTOT * CC * 2;
  __bf16* Kb  = (__bf16*)w; w += (size_t)MTOT * CC * 2;
  __bf16* Vb  = (__bf16*)w; w += (size_t)MTOT * CC * 2;
  __bf16* Xb  = (__bf16*)w; w += (size_t)MTOT * CC * 2;
  __bf16* Ab  = (__bf16*)w; w += (size_t)MTOT * KK * 2;
  __bf16* WqT = (__bf16*)w; w += (size_t)C3 * CC * 2;
  __bf16* WpT = (__bf16*)w; w += (size_t)CC * CC * 2;
  float*  kcf = (float*)w;  w += (size_t)BB * KK * CC * 4;
  float*  vcf = (float*)w;  w += (size_t)BB * KK * CC * 4;
  __bf16* kcb = (__bf16*)w; w += (size_t)BB * KK * CC * 2;
  __bf16* vcb = (__bf16*)w; w += (size_t)BB * KK * CC * 2;
  float*  dinv = (float*)w; w += (size_t)BB * KK * 4;
  (void)in_sizes; (void)n_in; (void)out_size; (void)ws_size;

  k_prep   <<<dim3(8192),        256, 0, stream>>>(A, X, Wq, Wp, Ab, Xi, WqT, WpT, kcf, vcf);
  k_denom  <<<dim3(KK, BB),      256, 0, stream>>>(A, dinv);
  k_qkv    <<<dim3(12, 256),     256, 0, stream>>>(Xi, WqT, Qb, Kb, Vb);
  k_pool   <<<dim3(16, BB, 16),   64, 0, stream>>>(Ab, Kb, Vb, kcf, vcf);
  k_finpool<<<dim3(256),         256, 0, stream>>>(kcf, vcf, dinv, kcb, vcb);
  k_attn   <<<dim3(128, HH, BB), 256, 0, stream>>>(Qb, Ab, kcb, vcb, cbias, Xb);
  k_proj   <<<dim3(4, 256),      256, 0, stream>>>(Xb, WpT, bp, out);
}
